// SelfAttention_85761906967138
// MI455X (gfx1250) — compile-verified, matrix operand order checked by reading, not run on hardware
//
#include <hip/hip_runtime.h>
#include <hip/hip_bf16.h>

typedef __bf16 bf16;
typedef __attribute__((ext_vector_type(16))) __bf16 v16bf;
typedef __attribute__((ext_vector_type(8)))  float  v8f;
typedef __attribute__((ext_vector_type(4)))  unsigned int u32x4;
typedef __attribute__((ext_vector_type(8)))  int i32x8;
typedef __attribute__((ext_vector_type(4)))  int i32x4;

#define BB  4
#define LLn 1024
#define HIDn 1024
#define HHn 16
#define DDn 64
#define BHn (BB*HHn)   // 64
#define MMn (BB*LLn)   // 4096

#if defined(__has_builtin)
#if __has_builtin(__builtin_amdgcn_tensor_load_to_lds) && __has_builtin(__builtin_amdgcn_s_wait_tensorcnt)
#define HAVE_TDM 1
#endif
#endif

union FragBF { v16bf v; uint4 q2[2]; bf16 h[16]; };
union FragF  { v8f  v; float f[8]; };

__device__ __forceinline__ bf16 f2bf(float x) { return (bf16)x; }

__device__ __forceinline__ v8f zero8() {
  v8f z = {0.f,0.f,0.f,0.f,0.f,0.f,0.f,0.f};
  return z;
}

__device__ __forceinline__ v8f wmma_bf16(const FragBF& a, const FragBF& b, v8f c) {
  return __builtin_amdgcn_wmma_f32_16x16x32_bf16(
      /*neg_a=*/false, a.v, /*neg_b=*/false, b.v,
      /*c_mod=*/(short)0, c, /*reuse_a=*/false, /*reuse_b=*/false);
}

// 16x32 bf16 A-fragment (or B-fragment of an [N x K] row-major source).
// Lane layout (ISA 7.12.2): lanes 0-15 row=lane, ks {0..7,16..23};
// lanes 16-31 row=lane-16, ks {8..15,24..31} -> two contiguous 16B chunks.
__device__ __forceinline__ FragBF frag_ld(const bf16* p0, size_t ld) {
  const int lane = threadIdx.x & 31;
  const bf16* p = p0 + (size_t)(lane & 15) * ld + ((lane >> 4) << 3);
  FragBF f;
  f.q2[0] = *(const uint4*)p;
  f.q2[1] = *(const uint4*)(p + 16);
  return f;
}

__device__ __forceinline__ FragBF frag_ld_f32(const float* p0, size_t ld) {
  const int lane = threadIdx.x & 31;
  const float* p = p0 + (size_t)(lane & 15) * ld + ((lane >> 4) << 3);
  float4 u0 = *(const float4*)(p);
  float4 u1 = *(const float4*)(p + 4);
  float4 u2 = *(const float4*)(p + 16);
  float4 u3 = *(const float4*)(p + 20);
  FragBF f;
  f.h[0]  = f2bf(u0.x); f.h[1]  = f2bf(u0.y); f.h[2]  = f2bf(u0.z); f.h[3]  = f2bf(u0.w);
  f.h[4]  = f2bf(u1.x); f.h[5]  = f2bf(u1.y); f.h[6]  = f2bf(u1.z); f.h[7]  = f2bf(u1.w);
  f.h[8]  = f2bf(u2.x); f.h[9]  = f2bf(u2.y); f.h[10] = f2bf(u2.z); f.h[11] = f2bf(u2.w);
  f.h[12] = f2bf(u3.x); f.h[13] = f2bf(u3.y); f.h[14] = f2bf(u3.z); f.h[15] = f2bf(u3.w);
  return f;
}

#ifdef HAVE_TDM
// One 2D TDM descriptor: copy tile_d1 rows x tile_d0 elements from global
// (row pitch stride_elems) into LDS at lds_off, packed row-major.
// dsz: 1 = 2-byte elements, 2 = 4-byte elements.
// D# layout per CDNA5 ISA 8.3/8.4; groups 2/3 unused for 2D tiles.
__device__ __forceinline__ void tdm_load_2d(unsigned lds_off, const void* gaddr,
                                            unsigned tile_d0, unsigned tile_d1,
                                            unsigned stride_elems, unsigned dsz) {
  unsigned long long ga = (unsigned long long)gaddr;
  u32x4 g0;
  g0.x = 1u;                                                // count=1, user mode
  g0.y = lds_off;                                           // lds_addr
  g0.z = (unsigned)(ga & 0xffffffffu);                      // global_addr[31:0]
  g0.w = (unsigned)((ga >> 32) & 0x01ffffffu) | (2u << 30); // [56:32] | type=2
  i32x8 g1;
  g1[0] = (int)(dsz << 16);                                 // data_size
  g1[1] = (int)((tile_d0 & 0xffffu) << 16);                 // tensor_dim0[15:0]
  g1[2] = (int)(((tile_d0 >> 16) & 0xffffu) | ((tile_d1 & 0xffffu) << 16));
  g1[3] = (int)(((tile_d1 >> 16) & 0xffffu) | ((tile_d0 & 0xffffu) << 16)); // | tile_dim0
  g1[4] = (int)(tile_d1 & 0xffffu);                         // tile_dim1 (dim2=0)
  g1[5] = (int)stride_elems;                                // tensor_dim0_stride[31:0]
  g1[6] = 0;
  g1[7] = 0;
  i32x4 gz = {0, 0, 0, 0};
#if defined(__clang_major__) && __clang_major__ >= 23
  i32x8 gz8 = {0, 0, 0, 0, 0, 0, 0, 0};
  __builtin_amdgcn_tensor_load_to_lds(g0, g1, gz, gz, gz8, 0);
#else
  __builtin_amdgcn_tensor_load_to_lds(g0, g1, gz, gz, 0);
#endif
}
#endif

// ---------------------------------------------------------------- convert
__global__ void k_f32_to_bf16(const float* __restrict__ s, bf16* __restrict__ d, int n) {
  for (int i = blockIdx.x * blockDim.x + threadIdx.x; i < n; i += gridDim.x * blockDim.x)
    d[i] = f2bf(s[i]);
}

// ---------------------------------------------------------------- QKV GEMM
// grid (MM/256, H, 3), block 256 (8 waves). Wave computes 32 rows x 64 cols.
// A tile (256 x 32 bf16) double-buffered in LDS, streamed by the TDM so the
// DMA of step i+1 overlaps the WMMA work of step i.
__global__ void __launch_bounds__(256)
k_qkv(const bf16* __restrict__ hbf,
      const bf16* __restrict__ wq, const bf16* __restrict__ wk, const bf16* __restrict__ wv,
      const float* __restrict__ bq, const float* __restrict__ bk, const float* __restrict__ bv,
      bf16* __restrict__ qo, bf16* __restrict__ ko, bf16* __restrict__ vo) {
  __shared__ bf16 atile[2][256 * 32];   // 2 x 16 KB
  const int wave = threadIdx.x >> 5;
  const int lane = threadIdx.x & 31;
  const int m0 = blockIdx.x * 256;
  const int h  = blockIdx.y;
  const int n0 = h * DDn;

  const bf16* w; const float* bias; bf16* dst;
  if (blockIdx.z == 0)      { w = wq; bias = bq; dst = qo; }
  else if (blockIdx.z == 1) { w = wk; bias = bk; dst = ko; }
  else                      { w = wv; bias = bv; dst = vo; }

  FragF c[2][4];
#pragma unroll
  for (int mt = 0; mt < 2; ++mt)
#pragma unroll
    for (int nt = 0; nt < 4; ++nt) c[mt][nt].v = zero8();

  const int NSTEP = HIDn / 32;
#ifdef HAVE_TDM
  if (wave == 0)
    tdm_load_2d((unsigned)(uintptr_t)&atile[0][0], hbf + (size_t)m0 * HIDn,
                /*tile_d0=*/32, /*tile_d1=*/256, /*stride=*/HIDn, /*dsz=*/1);
#endif

  for (int step = 0; step < NSTEP; ++step) {
    const int kk = step * 32;
#ifdef HAVE_TDM
    if (wave == 0) {
      if (step + 1 < NSTEP) {
        tdm_load_2d((unsigned)(uintptr_t)&atile[(step + 1) & 1][0],
                    hbf + (size_t)m0 * HIDn + kk + 32,
                    32, 256, HIDn, 1);
        __builtin_amdgcn_s_wait_tensorcnt(1);  // tile `step` resident
      } else {
        __builtin_amdgcn_s_wait_tensorcnt(0);
      }
    }
#else
    {
      const uint4* s4 = (const uint4*)(hbf + (size_t)(m0 + threadIdx.x) * HIDn + kk);
      uint4* d4 = (uint4*)(&atile[step & 1][0] + threadIdx.x * 32);
      d4[0] = s4[0]; d4[1] = s4[1]; d4[2] = s4[2]; d4[3] = s4[3];
    }
#endif
    __syncthreads();

    const bf16* at = &atile[step & 1][0];
    FragBF a0 = frag_ld(at + (wave * 32 + 0) * 32, 32);
    FragBF a1 = frag_ld(at + (wave * 32 + 16) * 32, 32);
#pragma unroll
    for (int nt = 0; nt < 4; ++nt) {
      FragBF b = frag_ld(w + (size_t)(n0 + nt * 16) * HIDn + kk, HIDn);
      c[0][nt].v = wmma_bf16(a0, b, c[0][nt].v);
      c[1][nt].v = wmma_bf16(a1, b, c[1][nt].v);
    }
    __syncthreads();   // all reads of this buffer done before step+2 overwrites it
  }

  const int col = lane & 15;
  const int rowoff = (lane >> 4) << 3;
#pragma unroll
  for (int mt = 0; mt < 2; ++mt) {
#pragma unroll
    for (int nt = 0; nt < 4; ++nt) {
      const int d = nt * 16 + col;
      const float bv_ = bias[n0 + d];
#pragma unroll
      for (int i = 0; i < 8; ++i) {
        const int m = m0 + wave * 32 + mt * 16 + i + rowoff;
        const int b = m >> 10, l = m & (LLn - 1);
        const int bh = b * HHn + h;
        dst[((size_t)bh * LLn + l) * DDn + d] = f2bf(c[mt][nt].f[i] + bv_);
      }
    }
  }
}

// ---------------------------------------------------------------- scores
// grid (L/16, L/16), block 256 (8 waves). Per block: one 16(l) x 16(r) tile
// for ALL 64 (b,h). 192KB dynamic LDS:
//   smF (64K f32 struct tile, TDM-streamed) + sq (32K) + sk (32K) + acc (64K).
// Phase order hides the struct-tile HBM stream behind term1 (q.k^T) WMMAs.
__global__ void __launch_bounds__(256)
k_scores(const float* __restrict__ sm, const bf16* __restrict__ wsq, const bf16* __restrict__ wsk,
         const float* __restrict__ bsq, const float* __restrict__ bsk,
         const bf16* __restrict__ qb, const bf16* __restrict__ kb,
         const float* __restrict__ mask, float* __restrict__ scores) {
  extern __shared__ char smem[];
  float* smF = (float*)smem;               // [16 l][16 r][64 d] f32
  bf16*  sqL = (bf16*)(smem + 65536);      // [256 pairs][64 d]
  bf16*  skL = (bf16*)(smem + 98304);      // [256 pairs][64 d]
  float* acc = (float*)(smem + 131072);    // [16 l][16 r][64 bh]

  const int wave = threadIdx.x >> 5;
  const int lane = threadIdx.x & 31;
  const int col = lane & 15;
  const int rowoff = (lane >> 4) << 3;
  const int l0 = blockIdx.x * 16;
  const int r0 = blockIdx.y * 16;
  const size_t LD = (size_t)LLn * DDn;

  // ---- Kick off the structure-tile DMA (16 rows of 16x64 f32, pitch L*D)
#ifdef HAVE_TDM
  if (wave == 0)
    tdm_load_2d((unsigned)(uintptr_t)smF,
                sm + ((size_t)l0 * LLn + r0) * DDn,
                /*tile_d0=*/16 * DDn, /*tile_d1=*/16,
                /*stride=*/LLn * DDn, /*dsz=*/2);
#else
  for (int e = threadIdx.x; e < 16 * 16 * DDn; e += 256) {
    const int l = e >> 10;                 // 16*64 = 1024 elements per l
    const int rd = e & 1023;
    smF[e] = sm[((size_t)(l0 + l) * LLn + r0) * DDn + rd];
  }
#endif

  // ---- Phase B1 (overlapped with TDM): term1 = q . k^T per bh -> acc
  for (int j = 0; j < 8; ++j) {
    const int bh = j * 8 + wave;
    const bf16* qp = qb + (size_t)bh * LD + (size_t)l0 * DDn;
    const bf16* kp = kb + (size_t)bh * LD + (size_t)r0 * DDn;
    FragBF a0 = frag_ld(qp + 0,  DDn), a1 = frag_ld(qp + 32, DDn);
    FragBF b0 = frag_ld(kp + 0,  DDn), b1 = frag_ld(kp + 32, DDn);
    FragF c; c.v = wmma_bf16(a0, b0, zero8()); c.v = wmma_bf16(a1, b1, c.v);
#pragma unroll
    for (int i = 0; i < 8; ++i)
      acc[((i + rowoff) * 16 + col) * 64 + bh] = c.f[i];   // [l][r][bh]
  }
#ifdef HAVE_TDM
  if (wave == 0) __builtin_amdgcn_s_wait_tensorcnt(0);
#endif
  __syncthreads();

  // ---- Phase A: sq/sk projection of the LDS struct tile (f32 -> bf16 WMMA)
  for (int mt = 0; mt < 2; ++mt) {
    const int lq = wave * 2 + mt;                          // local l, rows are r
    const float* sp = smF + lq * 16 * DDn;
    FragBF a0 = frag_ld_f32(sp + 0,  DDn);
    FragBF a1 = frag_ld_f32(sp + 32, DDn);
#pragma unroll
    for (int nt = 0; nt < 4; ++nt) {
      const int n = nt * 16 + col;
      {
        FragBF b0 = frag_ld(wsq + (size_t)(nt * 16) * DDn + 0,  DDn);
        FragBF b1 = frag_ld(wsq + (size_t)(nt * 16) * DDn + 32, DDn);
        FragF c; c.v = wmma_bf16(a0, b0, zero8()); c.v = wmma_bf16(a1, b1, c.v);
        const float bia = bsq[n];
#pragma unroll
        for (int i = 0; i < 8; ++i)
          sqL[(lq * 16 + i + rowoff) * 64 + n] = f2bf(c.f[i] + bia);
      }
      {
        FragBF b0 = frag_ld(wsk + (size_t)(nt * 16) * DDn + 0,  DDn);
        FragBF b1 = frag_ld(wsk + (size_t)(nt * 16) * DDn + 32, DDn);
        FragF c; c.v = wmma_bf16(a0, b0, zero8()); c.v = wmma_bf16(a1, b1, c.v);
        const float bia = bsk[n];
#pragma unroll
        for (int i = 0; i < 8; ++i)
          skL[(lq * 16 + i + rowoff) * 64 + n] = f2bf(c.f[i] + bia);
      }
    }
  }
  __syncthreads();

  // ---- Phase B2: term2 = sq[l] x q[:,l,:]  ->  [r x bh] per l
  for (int t = 0; t < 2; ++t) {
    const int l = wave * 2 + t;
    FragBF a0 = frag_ld(sqL + l * 16 * 64 + 0,  64);
    FragBF a1 = frag_ld(sqL + l * 16 * 64 + 32, 64);
#pragma unroll
    for (int nt = 0; nt < 4; ++nt) {
      const bf16* qp = qb + (size_t)(nt * 16) * LD + (size_t)(l0 + l) * DDn;
      FragBF b0 = frag_ld(qp + 0,  LD), b1 = frag_ld(qp + 32, LD);
      FragF c; c.v = wmma_bf16(a0, b0, zero8()); c.v = wmma_bf16(a1, b1, c.v);
#pragma unroll
      for (int i = 0; i < 8; ++i)
        acc[(l * 16 + i + rowoff) * 64 + nt * 16 + col] += c.f[i];
    }
  }
  __syncthreads();

  // ---- Phase B3: term3 = sk[:,r] x k[:,r,:]  ->  [l x bh] per r
  for (int t = 0; t < 2; ++t) {
    const int r = wave * 2 + t;
    FragBF a0 = frag_ld(skL + r * 64 + 0,  16 * 64);
    FragBF a1 = frag_ld(skL + r * 64 + 32, 16 * 64);
#pragma unroll
    for (int nt = 0; nt < 4; ++nt) {
      const bf16* kp = kb + (size_t)(nt * 16) * LD + (size_t)(r0 + r) * DDn;
      FragBF b0 = frag_ld(kp + 0,  LD), b1 = frag_ld(kp + 32, LD);
      FragF c; c.v = wmma_bf16(a0, b0, zero8()); c.v = wmma_bf16(a1, b1, c.v);
#pragma unroll
      for (int i = 0; i < 8; ++i)
        acc[((i + rowoff) * 16 + r) * 64 + nt * 16 + col] += c.f[i];
    }
  }
  __syncthreads();

  // ---- Phase C: scale + mask, write scores f32
  const float scale = 0.125f;  // 1/sqrt(64)
  for (int idx = threadIdx.x; idx < 16 * 16 * 64; idx += 256) {
    const int r = idx & 15, l = (idx >> 4) & 15, bh = idx >> 8;
    const int b = bh >> 4;
    const float v = acc[(l * 16 + r) * 64 + bh] * scale + mask[b * LLn + r0 + r];
    scores[((size_t)bh * LLn + (l0 + l)) * LLn + (r0 + r)] = v;
  }
}

// ---------------------------------------------------------------- softmax + PV
// grid (BH, L/128), block 256 (8 waves). Wave owns 16 rows; K-loop over r.
__global__ void __launch_bounds__(256)
k_softmax_pv(const float* __restrict__ scores, const bf16* __restrict__ vb,
             float* __restrict__ ctx) {
  __shared__ bf16 vt[64 * 32];   // V tile transposed: [d][r]
  const int wave = threadIdx.x >> 5;
  const int lane = threadIdx.x & 31;
  const int col = lane & 15;
  const int rowoff = (lane >> 4) << 3;
  const int g = rowoff;
  const int half = lane >> 4;
  const int bh = blockIdx.x;
  const int lt0 = blockIdx.y * 128;
  const int row_g = lt0 + wave * 16 + (lane & 15);

  const float* srow = scores + ((size_t)bh * LLn + row_g) * LLn;

  // pass 1: row max and row sum-of-exp (two lanes per row, halves combined)
  float m = -1e30f;
  const float4* s4 = (const float4*)(srow + half * 512);
#pragma unroll 4
  for (int j = 0; j < 128; ++j) {
    float4 v = s4[j];
    m = fmaxf(m, fmaxf(fmaxf(v.x, v.y), fmaxf(v.z, v.w)));
  }
  m = fmaxf(m, __shfl_xor(m, 16));
  float s = 0.f;
#pragma unroll 4
  for (int j = 0; j < 128; ++j) {
    float4 v = s4[j];
    s += __expf(v.x - m) + __expf(v.y - m) + __expf(v.z - m) + __expf(v.w - m);
  }
  s += __shfl_xor(s, 16);
  const float inv = 1.f / s;

  FragF c[4];
#pragma unroll
  for (int nt = 0; nt < 4; ++nt) c[nt].v = zero8();

  const bf16* vsrc = vb + (size_t)bh * LLn * DDn;
  for (int step = 0; step < 32; ++step) {
    const int r0 = step * 32;
    for (int e = threadIdx.x; e < 32 * 64; e += 256) {
      const int r = e >> 6, d = e & 63;
      vt[d * 32 + r] = vsrc[(size_t)(r0 + r) * DDn + d];
    }
    __syncthreads();

    FragBF a;
    const float* p = srow + r0 + g;
#pragma unroll
    for (int j = 0; j < 8; ++j) {
      a.h[j]     = f2bf(__expf(p[j]      - m) * inv);
      a.h[j + 8] = f2bf(__expf(p[16 + j] - m) * inv);
    }
#pragma unroll
    for (int nt = 0; nt < 4; ++nt) {
      FragBF b = frag_ld(vt + (size_t)(nt * 16) * 32, 32);
      c[nt].v = wmma_bf16(a, b, c[nt].v);
    }
    __syncthreads();
  }

  const int b = bh >> 4, h = bh & 15;
#pragma unroll
  for (int nt = 0; nt < 4; ++nt) {
    const int d = nt * 16 + col;
#pragma unroll
    for (int i = 0; i < 8; ++i) {
      const int l = lt0 + wave * 16 + i + rowoff;
      ctx[((size_t)b * LLn + l) * HIDn + h * DDn + d] = c[nt].f[i];
    }
  }
}

// ---------------------------------------------------------------- host
extern "C" void kernel_launch(void* const* d_in, const int* in_sizes, int n_in,
                              void* d_out, int out_size, void* d_ws, size_t ws_size,
                              hipStream_t stream) {
  (void)in_sizes; (void)n_in; (void)out_size; (void)ws_size;
  const float* hidden = (const float*)d_in[0];
  const float* mask   = (const float*)d_in[1];
  const float* sm     = (const float*)d_in[2];
  const float* Wq     = (const float*)d_in[3];
  const float* bq     = (const float*)d_in[4];
  const float* Wk     = (const float*)d_in[5];
  const float* bk     = (const float*)d_in[6];
  const float* Wv     = (const float*)d_in[7];
  const float* bv     = (const float*)d_in[8];
  const float* Wsq    = (const float*)d_in[9];
  const float* bsq    = (const float*)d_in[10];
  const float* Wsk    = (const float*)d_in[11];
  const float* bsk    = (const float*)d_in[12];

  float* ctx    = (float*)d_out;
  float* scores = (float*)d_out + (size_t)BB * LLn * HHn * DDn;

  char* ws = (char*)d_ws;
  bf16* hbf  = (bf16*)ws;  ws += (size_t)MMn * HIDn * 2;           // 8 MB
  bf16* wqb  = (bf16*)ws;  ws += (size_t)HIDn * HIDn * 2;          // 2 MB
  bf16* wkb  = (bf16*)ws;  ws += (size_t)HIDn * HIDn * 2;
  bf16* wvb  = (bf16*)ws;  ws += (size_t)HIDn * HIDn * 2;
  bf16* wsqb = (bf16*)ws;  ws += (size_t)DDn * DDn * 2 + 256;
  bf16* wskb = (bf16*)ws;  ws += (size_t)DDn * DDn * 2 + 256;
  bf16* qbuf = (bf16*)ws;  ws += (size_t)BHn * LLn * DDn * 2;      // 8 MB
  bf16* kbuf = (bf16*)ws;  ws += (size_t)BHn * LLn * DDn * 2;
  bf16* vbuf = (bf16*)ws;  ws += (size_t)BHn * LLn * DDn * 2;

  k_f32_to_bf16<<<4096, 256, 0, stream>>>(hidden, hbf, MMn * HIDn);
  k_f32_to_bf16<<<1024, 256, 0, stream>>>(Wq, wqb, HIDn * HIDn);
  k_f32_to_bf16<<<1024, 256, 0, stream>>>(Wk, wkb, HIDn * HIDn);
  k_f32_to_bf16<<<1024, 256, 0, stream>>>(Wv, wvb, HIDn * HIDn);
  k_f32_to_bf16<<<16, 256, 0, stream>>>(Wsq, wsqb, DDn * DDn);
  k_f32_to_bf16<<<16, 256, 0, stream>>>(Wsk, wskb, DDn * DDn);

  k_qkv<<<dim3(MMn / 256, HHn, 3), 256, 0, stream>>>(
      hbf, wqb, wkb, wvb, bq, bk, bv, qbuf, kbuf, vbuf);

  k_scores<<<dim3(LLn / 16, LLn / 16), 256, 196608, stream>>>(
      sm, wsqb, wskb, bsq, bsk, qbuf, kbuf, mask, scores);

  k_softmax_pv<<<dim3(BHn, LLn / 128), 256, 0, stream>>>(scores, vbuf, ctx);
}
